// HomoGNN_27427661152327
// MI455X (gfx1250) — matrix operand order repaired and checked by reading, not compiled, not run on hardware
//
#include <hip/hip_runtime.h>
#include <hip/hip_bf16.h>

// Problem constants (match reference)
#define NN 100000
#define EE 1600000
#define DD 128
#define RR 7
#define BB 4096
#define NSRC 8            // 7 relations + root
#define EPS 1e-5f

typedef __attribute__((ext_vector_type(16))) __bf16 v16bf;
typedef __attribute__((ext_vector_type(8)))  float  v8f;

// ---------------------------------------------------------------------------
// 1. h0[n,d] = embed[x[n], d]
__global__ void k_embed(const int* __restrict__ x, const float* __restrict__ embed,
                        float* __restrict__ h) {
  long long i = (long long)blockIdx.x * blockDim.x + threadIdx.x;
  if (i < (long long)NN * DD) {
    int n = (int)(i / DD), d = (int)(i % DD);
    h[i] = embed[(long long)x[n] * DD + d];
  }
}

// 2. cnt[dst*R + et] += 1  (float accumulate)
__global__ void k_count(const int* __restrict__ ei, const int* __restrict__ et,
                        float* __restrict__ cnt) {
  int e = blockIdx.x * blockDim.x + threadIdx.x;
  if (e < EE) {
    int dst = ei[EE + e];
    atomicAdd(&cnt[dst * RR + et[e]], 1.0f);
  }
}

// 3. cnt -> 1/max(cnt,1)
__global__ void k_invcnt(float* __restrict__ cnt) {
  int i = blockIdx.x * blockDim.x + threadIdx.x;
  if (i < NN * RR) cnt[i] = 1.0f / fmaxf(cnt[i], 1.0f);
}

// 4. pack all layer weights to bf16, TRANSPOSED: wbf[l][s][n][k] = W_s[k][n]
//    (s<7 -> rel_w[l][s], s==7 -> root_w[l]).  Column-major-for-B layout makes
//    every B fragment a contiguous 32-byte run -> global_load_b128 pairs.
__global__ void k_wconv(const float* __restrict__ rel_w, const float* __restrict__ root_w,
                        __bf16* __restrict__ wbf) {
  int i = blockIdx.x * blockDim.x + threadIdx.x;
  const int per_layer = NSRC * DD * DD;
  if (i >= 3 * per_layer) return;
  int l = i / per_layer;
  int rem = i % per_layer;
  int s = rem / (DD * DD);
  int nk = rem % (DD * DD);
  int n = nk / DD;          // output column
  int k = nk % DD;          // input (K) index
  float v = (s < RR) ? rel_w[(((long long)l * RR + s) * DD + k) * DD + n]
                     : root_w[((long long)l * DD + k) * DD + n];
  wbf[i] = (__bf16)v;
}

// 5. scatter: agg[(dst*R+et)*D + :] += h[src, :]   (1 wave per edge, 4 floats/lane)
__global__ void k_scatter(const int* __restrict__ ei, const int* __restrict__ et,
                          const float* __restrict__ h, float* __restrict__ agg) {
  long long t = (long long)blockIdx.x * blockDim.x + threadIdx.x;
  int e = (int)(t >> 5);
  int lane = (int)(t & 31);
  if (e >= EE) return;
  int src = ei[e];
  int dst = ei[EE + e];
  int r = et[e];
  const float* hs = h + (long long)src * DD + lane * 4;
  float* ag = agg + ((long long)dst * RR + r) * DD + lane * 4;
  float4 v = *(const float4*)hs;
  atomicAdd(ag + 0, v.x);
  atomicAdd(ag + 1, v.y);
  atomicAdd(ag + 2, v.z);
  atomicAdd(ag + 3, v.w);
}

// ---------------------------------------------------------------------------
// 6. WMMA transform: out[16,128] = sum_{s<8} Ahat_s[16,128] @ W_s[128,128]  (+bias,BN,ReLU)
//    Block = 16 nodes, 256 threads = 8 waves; wave w owns output columns [16w,16w+16).
__global__ __launch_bounds__(256)
void k_rgcn_transform(const float* __restrict__ agg, const float* __restrict__ cntinv,
                      const float* __restrict__ hin, const __bf16* __restrict__ wbf,
                      const float* __restrict__ bias,
                      const float* __restrict__ gamma, const float* __restrict__ beta,
                      const float* __restrict__ mean, const float* __restrict__ var,
                      int do_bn, float* __restrict__ hout) {
  __shared__ __bf16 sA[16][NSRC * DD + 8];   // 16 x 1032 bf16, padded vs bank conflicts
  const int tid = threadIdx.x;
  const int nodeBase = blockIdx.x * 16;

  // Stage A-tile into LDS as bf16: Ahat[m][s*128+k] = agg*invcnt (s<7) or h (s==7)
  for (int idx = tid; idx < 16 * NSRC * DD; idx += 256) {
    int m = idx >> 10;          // node row 0..15  (NSRC*DD == 1024)
    int kg = idx & 1023;        // 0..1023
    int s = kg >> 7;
    int k = kg & 127;
    int node = nodeBase + m;
    float v;
    if (s < RR) {
      long long seg = (long long)node * RR + s;
      v = agg[seg * DD + k] * cntinv[seg];
    } else {
      v = hin[(long long)node * DD + k];
    }
    sA[m][kg] = (__bf16)v;
  }
  __syncthreads();

  const int wave = tid >> 5;
  const int lane = tid & 31;
  const int m16 = lane & 15;
  const int hi  = lane >> 4;
  const int nBase = wave * 16;
  const int ncol = nBase + m16;

  v8f c = {};
  for (int s = 0; s < NSRC; ++s) {
    // Transposed weight slab for this source: Wt[n][k], our lane's column row:
    const __bf16* Wt = wbf + (size_t)s * DD * DD + (size_t)ncol * DD;
    __builtin_prefetch(Wt + DD * DD, 0, 0);   // next source's slab
    for (int kb = 0; kb < 4; ++kb) {          // 4 x K=32 chunks cover K=128
      const int kOff = s * DD + kb * 32;
      // A fragment (16x32 bf16): elem e -> kLocal = hi*16 + e (matches B's K order)
      //   -> two contiguous 8-element (16B) runs: ds_load_b128 pairs
      v16bf a;
      #pragma unroll
      for (int e = 0; e < 16; ++e) {
        int kl = hi * 16 + e;
        a[e] = sA[m16][kOff + kl];
      }
      // B fragment (32x16): lanes 0-15 K=0..15, lanes 16-31 K=16..31, column = ncol.
      // With transposed weights this is one aligned 32-byte contiguous load.
      v16bf b = *(const v16bf*)(Wt + kb * 32 + hi * 16);
      c = __builtin_amdgcn_wmma_f32_16x16x32_bf16(false, a, false, b,
                                                  (short)0, c, false, false);
    }
  }

  // Epilogue (C/D layout: VGPR v -> M = v + 8*hi, N = ncol)
  const float bcol = bias[ncol];
  float bn_scale = 1.0f, bn_shift = 0.0f;
  if (do_bn) {
    bn_scale = rsqrtf(var[ncol] + EPS) * gamma[ncol];
    bn_shift = beta[ncol] - mean[ncol] * bn_scale;
  }
  #pragma unroll
  for (int v = 0; v < 8; ++v) {
    int row = v + hi * 8;
    int node = nodeBase + row;
    float val = c[v] + bcol;
    if (do_bn) val = fmaxf(val * bn_scale + bn_shift, 0.0f);
    hout[(long long)node * DD + ncol] = val;
  }
}

// ---------------------------------------------------------------------------
// 7. Head: fc0 (gather home/away + 256-wide), fc1, fc2 + log_softmax
__global__ void k_fc0(const float* __restrict__ h, const int* __restrict__ home,
                      const int* __restrict__ away, const float* __restrict__ w,
                      const float* __restrict__ bvec, float* __restrict__ g0) {
  int b = blockIdx.x;
  int j = threadIdx.x;                 // 256 outputs
  const float* hh = h + (long long)home[b] * DD;
  const float* ha = h + (long long)away[b] * DD;
  const float* wr = w + j * 256;
  float acc = bvec[j];
  for (int i = 0; i < DD; ++i) acc += hh[i] * wr[i];
  for (int i = 0; i < DD; ++i) acc += ha[i] * wr[DD + i];
  g0[(long long)b * 256 + j] = fmaxf(acc, 0.0f);
}

__global__ void k_fc1(const float* __restrict__ g0, const float* __restrict__ w,
                      const float* __restrict__ bvec, float* __restrict__ g1) {
  int b = blockIdx.x;
  int j = threadIdx.x;                 // 128 outputs
  const float* gr = g0 + (long long)b * 256;
  const float* wr = w + j * 256;
  float acc = bvec[j];
  for (int i = 0; i < 256; ++i) acc += gr[i] * wr[i];
  g1[(long long)b * 128 + j] = fmaxf(acc, 0.0f);
}

__global__ void k_fc2(const float* __restrict__ g1, const float* __restrict__ w,
                      const float* __restrict__ bvec, float* __restrict__ out) {
  int b = blockIdx.x * blockDim.x + threadIdx.x;
  if (b >= BB) return;
  const float* gr = g1 + (long long)b * 128;
  float z[3];
  #pragma unroll
  for (int cc = 0; cc < 3; ++cc) {
    float acc = bvec[cc];
    const float* wr = w + cc * 128;
    for (int i = 0; i < 128; ++i) acc += gr[i] * wr[i];
    z[cc] = acc;
  }
  float mx = fmaxf(z[0], fmaxf(z[1], z[2]));
  float s = expf(z[0] - mx) + expf(z[1] - mx) + expf(z[2] - mx);
  float lse = mx + logf(s);
  out[(long long)b * 3 + 0] = z[0] - lse;
  out[(long long)b * 3 + 1] = z[1] - lse;
  out[(long long)b * 3 + 2] = z[2] - lse;
}

// ---------------------------------------------------------------------------
extern "C" void kernel_launch(void* const* d_in, const int* in_sizes, int n_in,
                              void* d_out, int out_size, void* d_ws, size_t ws_size,
                              hipStream_t stream) {
  const int*   x      = (const int*)  d_in[0];
  const int*   ei     = (const int*)  d_in[1];   // [2,E]
  const int*   et     = (const int*)  d_in[2];   // [E]
  const int*   home   = (const int*)  d_in[3];
  const int*   away   = (const int*)  d_in[4];
  const float* embed  = (const float*)d_in[5];
  const float* rel_w  = (const float*)d_in[6];   // [3,R,D,D]
  const float* root_w = (const float*)d_in[7];   // [3,D,D]
  const float* conv_b = (const float*)d_in[8];   // [3,D]
  const float* bn_g   = (const float*)d_in[9];   // [2,D]
  const float* bn_b   = (const float*)d_in[10];
  const float* bn_m   = (const float*)d_in[11];
  const float* bn_v   = (const float*)d_in[12];
  const float* fc0_w  = (const float*)d_in[13];
  const float* fc0_b  = (const float*)d_in[14];
  const float* fc1_w  = (const float*)d_in[15];
  const float* fc1_b  = (const float*)d_in[16];
  const float* fc2_w  = (const float*)d_in[17];
  const float* fc2_b  = (const float*)d_in[18];
  float* out = (float*)d_out;

  // Workspace carve-up (f32 units)
  float* ws = (float*)d_ws;
  size_t off = 0;
  float* h0   = ws + off; off += (size_t)NN * DD;        // 12.8M
  float* h1   = ws + off; off += (size_t)NN * DD;        // 12.8M
  float* agg  = ws + off; off += (size_t)NN * RR * DD;   // 89.6M
  float* cnt  = ws + off; off += (size_t)NN * RR;        // 0.7M
  float* g0   = ws + off; off += (size_t)BB * 256;
  float* g1   = ws + off; off += (size_t)BB * 128;
  __bf16* wbf = (__bf16*)(ws + off);                     // 3*8*D*D bf16 (32B aligned)

  const int T = 256;

  // Counts (graph-invariant within a launch, recomputed every call)
  hipMemsetAsync(cnt, 0, (size_t)NN * RR * sizeof(float), stream);
  k_embed<<<(int)(((long long)NN * DD + T - 1) / T), T, 0, stream>>>(x, embed, h0);
  k_count<<<(EE + T - 1) / T, T, 0, stream>>>(ei, et, cnt);
  k_invcnt<<<(NN * RR + T - 1) / T, T, 0, stream>>>(cnt);
  k_wconv<<<(3 * NSRC * DD * DD + T - 1) / T, T, 0, stream>>>(rel_w, root_w, wbf);

  float* hin = h0;
  float* hout = h1;
  for (int l = 0; l < 3; ++l) {
    hipMemsetAsync(agg, 0, (size_t)NN * RR * DD * sizeof(float), stream);
    k_scatter<<<(int)(((long long)EE * 32 + T - 1) / T), T, 0, stream>>>(ei, et, hin, agg);
    int do_bn = (l < 2) ? 1 : 0;
    const float* g = do_bn ? bn_g + l * DD : bn_g;  // unused when do_bn==0
    const float* b = do_bn ? bn_b + l * DD : bn_b;
    const float* m = do_bn ? bn_m + l * DD : bn_m;
    const float* v = do_bn ? bn_v + l * DD : bn_v;
    k_rgcn_transform<<<NN / 16, 256, 0, stream>>>(
        agg, cnt, hin, wbf + (size_t)l * NSRC * DD * DD, conv_b + l * DD,
        g, b, m, v, do_bn, hout);
    float* tmp = hin; hin = hout; hout = tmp;
  }
  // after 3 swaps, final features live in `hin`

  k_fc0<<<BB, 256, 0, stream>>>(hin, home, away, fc0_w, fc0_b, g0);
  k_fc1<<<BB, 128, 0, stream>>>(g0, fc1_w, fc1_b, g1);
  k_fc2<<<(BB + T - 1) / T, T, 0, stream>>>(g1, fc2_w, fc2_b, out);
}
